// EdgeModel_17669495456023
// MI455X (gfx1250) — matrix operand order repaired and checked by reading, not compiled, not run on hardware
//
#include <hip/hip_runtime.h>

typedef __attribute__((ext_vector_type(16))) _Float16 v16h;
typedef __attribute__((ext_vector_type(8)))  _Float16 v8h;
typedef __attribute__((ext_vector_type(8)))  float    v8f;

#define NODE_IN 64
#define EDGE_IN 32
#define HID 128
#define OUTF 64
#define NWAVES 4
#define TILES_PER_WAVE 16

__global__ __launch_bounds__(NWAVES * 32) void edge_mlp_kernel(
    const float* __restrict__ src, const float* __restrict__ dst,
    const float* __restrict__ eattr,
    const float* __restrict__ W1, const float* __restrict__ b1,
    const float* __restrict__ W2, const float* __restrict__ b2,
    float* __restrict__ out, int E, int numTiles)
{
    // Weights pre-swizzled into WMMA B-fragment layout:
    // B 32x16 f16 fragment: lane l holds N = l%16, j in [0,16): K = (l/16)*16 + j
    __shared__ v16h sW1[40][32];                    // [t*5+c][lane]  40 KB
    __shared__ v16h sW2[16][32];                    // [t2*4+c2][lane] 16 KB
    __shared__ __align__(32) _Float16 sH[NWAVES][16][32];  // h staging, 4 KB

    const int tid = threadIdx.x;

    // ---- prologue: convert f32 weights -> f16 B-fragments in LDS ----
    for (int idx = tid; idx < 40 * 32; idx += NWAVES * 32) {
        const int f = idx >> 5, l = idx & 31;
        const int t = f / 5, c = f % 5;
        const int n = t * 16 + (l & 15);
        const int kbase = c * 32 + (l >> 4) * 16;
        _Float16* p = (_Float16*)&sW1[f][l];
#pragma unroll
        for (int j = 0; j < 16; ++j)
            p[j] = (_Float16)W1[(kbase + j) * HID + n];
    }
    for (int idx = tid; idx < 16 * 32; idx += NWAVES * 32) {
        const int f = idx >> 5, l = idx & 31;
        const int t = f >> 2, c = f & 3;
        const int n = t * 16 + (l & 15);
        const int kbase = c * 32 + (l >> 4) * 16;
        _Float16* p = (_Float16*)&sW2[f][l];
#pragma unroll
        for (int j = 0; j < 16; ++j)
            p[j] = (_Float16)W2[(kbase + j) * OUTF + n];
    }
    __syncthreads();

    const int wave  = tid >> 5;
    const int lane  = tid & 31;
    const int mrow  = lane & 15;   // M index (edge within tile) for A, N index for C/D
    const int khalf = lane >> 4;   // K-half selector for A, M-half for C/D

    // biases are tile-invariant per lane
    float b1v[8], b2v[4];
#pragma unroll
    for (int t = 0; t < 8; ++t) b1v[t] = b1[t * 16 + mrow];
#pragma unroll
    for (int t = 0; t < 4; ++t) b2v[t] = b2[t * 16 + mrow];

    const int tile0 = ((int)blockIdx.x * NWAVES + wave) * TILES_PER_WAVE;

    for (int ti = 0; ti < TILES_PER_WAVE; ++ti) {
        const int tile = tile0 + ti;
        if (tile >= numTiles) break;
        const long e0  = (long)tile * 16;
        long row       = e0 + mrow;
        const long rowc = row < E ? row : (long)(E - 1);

        // ---- build 5 A-fragments of x = [src | dest | edge_attr], f32 -> f16 ----
        // A 16x32 f16: lane l row M=l%16; j in [0,16): k_local = (l/16)*16 + j
        v16h xf[5];
#pragma unroll
        for (int c = 0; c < 5; ++c) {
            const float* p;
            if (c < 2)       p = src   + rowc * NODE_IN + c * 32;
            else if (c < 4)  p = dst   + rowc * NODE_IN + (c - 2) * 32;
            else             p = eattr + rowc * EDGE_IN;
            const float4* q0 = (const float4*)(p + khalf * 16);
            const float4* q1 = (const float4*)(p + khalf * 16 + 8);
            float4 f0 = q0[0], f1 = q0[1], f2 = q1[0], f3 = q1[1];
            v16h a;
            a[0]=(_Float16)f0.x;  a[1]=(_Float16)f0.y;  a[2]=(_Float16)f0.z;  a[3]=(_Float16)f0.w;
            a[4]=(_Float16)f1.x;  a[5]=(_Float16)f1.y;  a[6]=(_Float16)f1.z;  a[7]=(_Float16)f1.w;
            a[8]=(_Float16)f2.x;  a[9]=(_Float16)f2.y;  a[10]=(_Float16)f2.z; a[11]=(_Float16)f2.w;
            a[12]=(_Float16)f3.x; a[13]=(_Float16)f3.y; a[14]=(_Float16)f3.z; a[15]=(_Float16)f3.w;
            xf[c] = a;
        }

        // ---- out accumulators (16x64), init with b2 ----
        v8f oacc[4];
#pragma unroll
        for (int t2 = 0; t2 < 4; ++t2) {
#pragma unroll
            for (int r = 0; r < 8; ++r) oacc[t2][r] = b2v[t2];
        }

#pragma unroll
        for (int c2 = 0; c2 < 4; ++c2) {
            // hidden columns [c2*32, c2*32+32) = GEMM1 N-tiles 2*c2, 2*c2+1
#pragma unroll
            for (int s = 0; s < 2; ++s) {
                const int t = 2 * c2 + s;
                v8f hacc;
#pragma unroll
                for (int r = 0; r < 8; ++r) hacc[r] = b1v[t];
#pragma unroll
                for (int c = 0; c < 5; ++c) {
                    hacc = __builtin_amdgcn_wmma_f32_16x16x32_f16(
                        false, xf[c], false, sW1[t * 5 + c][lane],
                        (short)0, hacc, false, false);
                }
                // ReLU + stage h (f16) in accumulator layout: c[r] -> M=r+8*khalf, N=mrow
#pragma unroll
                for (int r = 0; r < 8; ++r) {
                    float v = hacc[r] > 0.f ? hacc[r] : 0.f;
                    sH[wave][r + 8 * khalf][s * 16 + mrow] = (_Float16)v;
                }
            }
            // reload staged h as A-fragment (wave-local; LDS ops are in-order per wave)
            v8h lo = *(const v8h*)&sH[wave][mrow][khalf * 16];
            v8h hi = *(const v8h*)&sH[wave][mrow][khalf * 16 + 8];
            v16h ha;
#pragma unroll
            for (int i = 0; i < 8; ++i) { ha[i] = lo[i]; ha[i + 8] = hi[i]; }
#pragma unroll
            for (int t2 = 0; t2 < 4; ++t2) {
                oacc[t2] = __builtin_amdgcn_wmma_f32_16x16x32_f16(
                    false, ha, false, sW2[t2 * 4 + c2][lane],
                    (short)0, oacc[t2], false, false);
            }
        }

        // ---- store out tile [16 x 64] f32 ----
#pragma unroll
        for (int r = 0; r < 8; ++r) {
            const long er = e0 + r + 8 * khalf;
            if (er < E) {
                float* po = out + er * OUTF + mrow;
#pragma unroll
                for (int t2 = 0; t2 < 4; ++t2) po[t2 * 16] = oacc[t2][r];
            }
        }
    }
}

extern "C" void kernel_launch(void* const* d_in, const int* in_sizes, int n_in,
                              void* d_out, int out_size, void* d_ws, size_t ws_size,
                              hipStream_t stream) {
    (void)n_in; (void)out_size; (void)d_ws; (void)ws_size;
    const float* src = (const float*)d_in[0];
    const float* dst = (const float*)d_in[1];
    const float* ea  = (const float*)d_in[2];
    // d_in[3] = u, d_in[4] = batch : unused by reference
    const float* W1  = (const float*)d_in[5];
    const float* b1  = (const float*)d_in[6];
    const float* W2  = (const float*)d_in[7];
    const float* b2  = (const float*)d_in[8];
    float* out = (float*)d_out;

    const int E = in_sizes[0] / NODE_IN;
    const int numTiles = (E + 15) / 16;
    const int tilesPerWG = NWAVES * TILES_PER_WAVE;
    const int numWG = (numTiles + tilesPerWG - 1) / tilesPerWG;
    edge_mlp_kernel<<<numWG, NWAVES * 32, 0, stream>>>(
        src, dst, ea, W1, b1, W2, b2, out, E, numTiles);
}
